// CausalSelfAttention_51848845197421
// MI455X (gfx1250) — matrix operand order repaired and checked by reading, not compiled, not run on hardware
//
#include <hip/hip_runtime.h>
#include <hip/hip_bf16.h>

#define B_    2
#define T_    2048
#define C_    1024
#define H_    16
#define HKV_  4
#define HD_   64
#define WSW_  1024

typedef __attribute__((ext_vector_type(16))) _Float16 v16h;
typedef __attribute__((ext_vector_type(8)))  _Float16 v8h;
typedef __attribute__((ext_vector_type(8)))  float    v8f;

__device__ __forceinline__ v8f zero_v8f() {
  v8f z = {0.f, 0.f, 0.f, 0.f, 0.f, 0.f, 0.f, 0.f};
  return z;
}

// A-matrix 16x32 f16 fragment (CDNA5 layout):
// lanes 0-15: row M=lane, halves 0..7 = K 0..7, halves 8..15 = K 16..23
// lanes 16-31: row M=lane-16, halves 0..7 = K 8..15, halves 8..15 = K 24..31
__device__ __forceinline__ v16h load_a_frag(const _Float16* base, int ld) {
  const int l = threadIdx.x & 31;
  const int row = l & 15;
  const int koff = (l < 16) ? 0 : 8;
  const _Float16* p = base + row * ld + koff;
  v8h lo = *(const v8h*)(p);
  v8h hi = *(const v8h*)(p + 16);
  v16h r;
#pragma unroll
  for (int i = 0; i < 8; ++i) { r[i] = lo[i]; r[i + 8] = hi[i]; }
  return r;
}

// B-matrix 32x16 f16 fragment: lane holds column N=lane&15;
// lanes 0-15: K=0..7,16..23, lanes 16-31: K=8..15,24..31 (Bt stored N x K row-major)
__device__ __forceinline__ v16h load_b_frag(const _Float16* base, int ld) {
  const int l = threadIdx.x & 31;
  const int col = l & 15;
  const int koff = (l < 16) ? 0 : 8;
  const _Float16* p = base + col * ld + koff;
  v8h lo = *(const v8h*)(p);
  v8h hi = *(const v8h*)(p + 16);
  v16h r;
#pragma unroll
  for (int i = 0; i < 8; ++i) { r[i] = lo[i]; r[i + 8] = hi[i]; }
  return r;
}

#define WMMA_F16(a, b, c) \
  __builtin_amdgcn_wmma_f32_16x16x32_f16(false, (a), false, (b), (short)0, (c), false, false)

// ---------------------------------------------------------------------------
// GEMM: C(MxN,f32) = A(MxK,f16 row-major) * Bt(NxK,f16 row-major)^T
// Block: 8 waves as 2(M) x 4(N); wave tile 32x64 (2x4 WMMA tiles).
// Ping-pong double buffering (2x-unrolled k-loop) -> no register rotation
// copies; loads for one buffer overlap WMMAs consuming the other.
// Requires K % 64 == 32.
// ---------------------------------------------------------------------------
struct Frags {
  v16h a0, a1;
  v16h b[4];
};

__device__ __forceinline__ void load_frags(Frags& f, const _Float16* arow0,
                                           const _Float16* arow1,
                                           const _Float16* bbase, int k0, int K) {
  f.a0 = load_a_frag(arow0 + k0, K);
  f.a1 = load_a_frag(arow1 + k0, K);
#pragma unroll
  for (int t = 0; t < 4; ++t)
    f.b[t] = load_b_frag(bbase + (size_t)(16 * t) * K + k0, K);
}

__device__ __forceinline__ void mma_frags(v8f acc[2][4], const Frags& f) {
#pragma unroll
  for (int t = 0; t < 4; ++t) {
    acc[0][t] = WMMA_F16(f.a0, f.b[t], acc[0][t]);
    acc[1][t] = WMMA_F16(f.a1, f.b[t], acc[1][t]);
  }
}

__global__ __launch_bounds__(256) void gemm_xWt_f16(
    const _Float16* __restrict__ A, const _Float16* __restrict__ Bt,
    float* __restrict__ C, int M, int N, int K)
{
  const int wave  = threadIdx.x >> 5;
  const int waveM = wave >> 2;       // 0..1
  const int waveN = wave & 3;        // 0..3
  const int m0 = blockIdx.x * 64 + waveM * 32;
  const int n0 = blockIdx.y * 256 + waveN * 64;

  v8f acc[2][4];
#pragma unroll
  for (int mi = 0; mi < 2; ++mi)
#pragma unroll
    for (int t = 0; t < 4; ++t) acc[mi][t] = zero_v8f();

  const _Float16* arow0 = A + (size_t)m0 * K;
  const _Float16* arow1 = A + (size_t)(m0 + 16) * K;
  const _Float16* bbase = Bt + (size_t)n0 * K;

  Frags f0, f1;
  load_frags(f0, arow0, arow1, bbase, 0, K);
  // steps at k = 0,32,...,K-32 (K/32 odd). Ping-pong pairs:
  for (int k0 = 32; k0 + 32 < K; k0 += 64) {
    load_frags(f1, arow0, arow1, bbase, k0, K);        // step k0
    mma_frags(acc, f0);                                // step k0-32
    load_frags(f0, arow0, arow1, bbase, k0 + 32, K);   // step k0+32
    mma_frags(acc, f1);                                // step k0
  }
  mma_frags(acc, f0);                                  // final step K-32

  // C layout: vgpr r -> row r + 8*(lane>=16); col = lane&15
  const int l = threadIdx.x & 31;
  const int col = l & 15;
  const int rbase = (l >> 4) * 8;
#pragma unroll
  for (int mi = 0; mi < 2; ++mi)
#pragma unroll
    for (int t = 0; t < 4; ++t)
#pragma unroll
      for (int r = 0; r < 8; ++r)
        C[(size_t)(m0 + 16 * mi + rbase + r) * N + n0 + 16 * t + col] = acc[mi][t][r];
}

// ---------------------------------------------------------------------------
// Converters
// ---------------------------------------------------------------------------
__global__ void cvt_f16(const float* __restrict__ in, _Float16* __restrict__ out, int n) {
  int i = blockIdx.x * 256 + threadIdx.x;
  if (i < n) out[i] = (_Float16)in[i];
}

__global__ void transpose_cvt_f16(const float* __restrict__ W, _Float16* __restrict__ Wt,
                                  int K, int N) {
  int i = blockIdx.x * 256 + threadIdx.x;
  if (i < K * N) {
    int k = i / N, n = i % N;
    Wt[(size_t)n * K + k] = (_Float16)W[i];
  }
}

// ---------------------------------------------------------------------------
// RoPE + RMSNorm: f32 (B,T) x ldrow (heads packed at +h*64) -> f16 packed
// (B,T,NH,64). One block of 64 threads per (b,t,h).
// ---------------------------------------------------------------------------
__global__ __launch_bounds__(64) void rope_rmsnorm_f16(
    const float* __restrict__ uin, int ldrow,
    const float* __restrict__ cosT, const float* __restrict__ sinT,
    _Float16* __restrict__ uout, int NH, float outScale)
{
  const int bth = blockIdx.x;
  const int h = bth % NH;
  const int bt = bth / NH;
  const int t = bt % T_;
  const int i = threadIdx.x;
  const float* u = uin + (size_t)bt * ldrow + h * HD_;
  float val;
  if (i < 32) {
    const float cs = cosT[t * 32 + i];
    const float sn = sinT[t * 32 + i];
    val = u[i] * cs + u[i + 32] * sn;
  } else {
    const int j = i - 32;
    const float cs = cosT[t * 32 + j];
    const float sn = sinT[t * 32 + j];
    val = -u[j] * sn + u[i] * cs;
  }
  __shared__ float s2[64];
  s2[i] = val * val;
  __syncthreads();
  if (i < 32) s2[i] += s2[i + 32];
  __syncthreads();
  if (i < 16) s2[i] += s2[i + 16];
  __syncthreads();
  if (i < 8) s2[i] += s2[i + 8];
  __syncthreads();
  if (i < 4) s2[i] += s2[i + 4];
  __syncthreads();
  if (i < 2) s2[i] += s2[i + 2];
  __syncthreads();
  if (i < 1) s2[i] += s2[i + 1];
  __syncthreads();
  const float rms = rsqrtf(s2[0] * (1.0f / 64.0f) + 1.1920929e-07f);
  uout[((size_t)bt * NH + h) * HD_ + i] = (_Float16)(val * rms * outScale);
}

// ---------------------------------------------------------------------------
// v = v + 2*sigmoid(x[:,:32] @ Wg[:,h]) * ve, written TRANSPOSED as
// vt (B,HKV,HD,T) f16 so P@V B-fragments are contiguous. Block per (b,t,h).
// ---------------------------------------------------------------------------
__global__ __launch_bounds__(64) void v_gate_transpose(
    const float* __restrict__ vin, int ldrow, const float* __restrict__ x,
    const float* __restrict__ ve, const float* __restrict__ Wg,
    _Float16* __restrict__ vt)
{
  const int idx = blockIdx.x;            // (b*T + t)*HKV + h
  const int h = idx & (HKV_ - 1);
  const int bt = idx >> 2;
  const int t = bt & (T_ - 1);
  const int b = bt >> 11;
  const int i = threadIdx.x;
  __shared__ float gsh;
  if (i < 32) {
    float p = x[(size_t)bt * C_ + i] * Wg[i * HKV_ + h];
#pragma unroll
    for (int off = 16; off > 0; off >>= 1) p += __shfl_xor(p, off, 32);
    if (i == 0) gsh = 2.0f / (1.0f + __expf(-p));
  }
  __syncthreads();
  const float g = gsh;
  const float v = vin[(size_t)bt * ldrow + h * HD_ + i] + g * ve[(size_t)idx * HD_ + i];
  vt[(((size_t)b * HKV_ + h) * HD_ + i) * T_ + t] = (_Float16)v;
}

// ---------------------------------------------------------------------------
// Sliding-window GQA flash attention. One wave per (b, head, 16-query tile).
// qh pre-scaled by 1/sqrt(HD). Pipelined: V frags loaded early in iteration,
// next block's K frags prefetched (in place, no rotation) behind score WMMAs.
// ---------------------------------------------------------------------------
__global__ __launch_bounds__(256) void attn_swa(
    const _Float16* __restrict__ qh,   // (B,T,H,HD)
    const _Float16* __restrict__ kh,   // (B,T,HKV,HD)
    const _Float16* __restrict__ vt,   // (B,HKV,HD,T)
    _Float16* __restrict__ yh)         // (B,T,H,HD)
{
  __shared__ __align__(16) _Float16 plds_all[8][16][32];
  const int wave = threadIdx.x >> 5;
  const int l = threadIdx.x & 31;
  const int hi = l >> 4;
  const int coll = l & 15;
  _Float16(*plds)[32] = plds_all[wave];

  const int w  = blockIdx.x * 8 + wave;  // 0..4095
  const int qt = w & 127;
  const int h  = (w >> 7) & 15;
  const int b  = w >> 11;
  const int kvh = h >> 2;
  const int q0 = qt * 16;

  const _Float16* qbase = qh + (((size_t)b * T_ + q0) * H_ + h) * HD_;
  const v16h qa0 = load_a_frag(qbase, H_ * HD_);        // channels 0..31
  const v16h qa1 = load_a_frag(qbase + 32, H_ * HD_);   // channels 32..63

  v8f o[4];
#pragma unroll
  for (int t = 0; t < 4; ++t) o[t] = zero_v8f();
  float mrow[8], lrow[8];
#pragma unroll
  for (int r = 0; r < 8; ++r) { mrow[r] = -1.0e30f; lrow[r] = 0.0f; }

  int jlo = q0 - WSW_ + 1;
  if (jlo < 0) jlo = 0;
  jlo &= ~31;                            // align down; extra keys get masked
  const int jend = q0 + 15;

  const int ldk = HKV_ * HD_;            // 256
  const _Float16* khead = kh + ((size_t)b * T_ * HKV_ + kvh) * HD_;
  const _Float16* vhead = vt + ((size_t)b * HKV_ + kvh) * HD_ * T_;

  // prologue: K fragments for the first 32-key block
  v16h kc[4];
  {
    const _Float16* kp = khead + (size_t)jlo * ldk;
    kc[0] = load_b_frag(kp, ldk);
    kc[1] = load_b_frag(kp + 32, ldk);
    kc[2] = load_b_frag(kp + 16 * ldk, ldk);
    kc[3] = load_b_frag(kp + 16 * ldk + 32, ldk);
  }

  for (int j0 = jlo; j0 <= jend; j0 += 32) {
    // V fragments for this block: issued early, consumed after softmax
    v16h vb[4];
#pragma unroll
    for (int t = 0; t < 4; ++t)
      vb[t] = load_b_frag(vhead + (size_t)(16 * t) * T_ + j0, T_);

    // ---- scores: two 16x16 tiles over HD=64 ----
    v8f s0 = zero_v8f(), s1 = zero_v8f();
    s0 = WMMA_F16(qa0, kc[0], s0);
    s0 = WMMA_F16(qa1, kc[1], s0);
    s1 = WMMA_F16(qa0, kc[2], s1);
    s1 = WMMA_F16(qa1, kc[3], s1);

    // prefetch next block's K fragments behind the score WMMAs
    if (j0 + 32 <= jend) {
      const _Float16* kp = khead + (size_t)(j0 + 32) * ldk;
      kc[0] = load_b_frag(kp, ldk);
      kc[1] = load_b_frag(kp + 32, ldk);
      kc[2] = load_b_frag(kp + 16 * ldk, ldk);
      kc[3] = load_b_frag(kp + 16 * ldk + 32, ldk);
    }

    // ---- mask + online softmax (row stats live in C layout) ----
    const int key0 = j0 + coll;
    const int key1 = j0 + 16 + coll;
    const float ninf = -__builtin_inff();
#pragma unroll
    for (int r = 0; r < 8; ++r) {
      const int q = q0 + r + 8 * hi;
      if (key0 > q || key0 < q - (WSW_ - 1)) s0[r] = ninf;
      if (key1 > q || key1 < q - (WSW_ - 1)) s1[r] = ninf;
    }
#pragma unroll
    for (int r = 0; r < 8; ++r) {
      float mv = fmaxf(s0[r], s1[r]);
#pragma unroll
      for (int off = 1; off < 16; off <<= 1)
        mv = fmaxf(mv, __shfl_xor(mv, off, 32));
      const float mnew = fmaxf(mrow[r], mv);          // never -inf (init -1e30)
      const float rescale = __expf(mrow[r] - mnew);
      mrow[r] = mnew;
      const float p0 = __expf(s0[r] - mnew);          // masked -> exp(-inf)=0
      const float p1 = __expf(s1[r] - mnew);
      float ps = p0 + p1;
#pragma unroll
      for (int off = 1; off < 16; off <<= 1)
        ps += __shfl_xor(ps, off, 32);
      lrow[r] = lrow[r] * rescale + ps;
#pragma unroll
      for (int t = 0; t < 4; ++t) o[t][r] *= rescale;
      plds[r + 8 * hi][coll] = (_Float16)p0;          // C layout -> LDS
      plds[r + 8 * hi][coll + 16] = (_Float16)p1;
    }
    // ---- P @ V: reload P as A-fragment, 4 N-tiles of HD ----
    const v16h pa = load_a_frag(&plds[0][0], 32);
#pragma unroll
    for (int t = 0; t < 4; ++t) o[t] = WMMA_F16(pa, vb[t], o[t]);
  }

  // ---- normalize + store f16 for the projection GEMM ----
#pragma unroll
  for (int r = 0; r < 8; ++r) {
    const float inv = 1.0f / lrow[r];                 // >= 1 (key q always valid)
    const int row = q0 + r + 8 * hi;
#pragma unroll
    for (int t = 0; t < 4; ++t)
      yh[(((size_t)b * T_ + row) * H_ + h) * HD_ + 16 * t + coll] =
          (_Float16)(o[t][r] * inv);
  }
}

// ---------------------------------------------------------------------------
extern "C" void kernel_launch(void* const* d_in, const int* in_sizes, int n_in,
                              void* d_out, int out_size, void* d_ws, size_t ws_size,
                              hipStream_t stream) {
  (void)in_sizes; (void)n_in; (void)out_size; (void)ws_size;
  const float* x    = (const float*)d_in[0];
  const float* ve   = (const float*)d_in[1];
  const float* cosT = (const float*)d_in[2];
  const float* sinT = (const float*)d_in[3];
  const float* Wq   = (const float*)d_in[4];
  const float* Wk   = (const float*)d_in[5];
  const float* Wv   = (const float*)d_in[6];
  const float* Wp   = (const float*)d_in[7];
  const float* Wg   = (const float*)d_in[8];
  float* out = (float*)d_out;

  const int M = B_ * T_;       // 4096
  const int NKV = HKV_ * HD_;  // 256
  const int NKVF = 2 * NKV;    // fused K|V projection width (512)

  char* ws = (char*)d_ws;
  size_t off = 0;
  auto alloc = [&](size_t bytes) -> char* {
    char* p = ws + off;
    off = (off + bytes + 255) & ~(size_t)255;
    return p;
  };
  _Float16* xh   = (_Float16*)alloc((size_t)M * C_ * 2);
  _Float16* Wqt  = (_Float16*)alloc((size_t)C_ * C_ * 2);
  _Float16* Wkvt = (_Float16*)alloc((size_t)NKVF * C_ * 2);  // rows 0..255 = K, 256..511 = V
  _Float16* Wpt  = (_Float16*)alloc((size_t)C_ * C_ * 2);
  float*    qf   = (float*)alloc((size_t)M * C_ * 4);
  float*    kvf  = (float*)alloc((size_t)M * NKVF * 4);      // per row: [K(256) | V(256)]
  _Float16* qhb  = (_Float16*)alloc((size_t)M * C_ * 2);
  _Float16* khb  = (_Float16*)alloc((size_t)M * NKV * 2);
  _Float16* vtb  = (_Float16*)alloc((size_t)M * NKV * 2);
  _Float16* yh   = xh;  // xh dead after the QKV GEMMs; reuse for attention out

  // Stage 0: precision conversion + weight transposes (Wt is N x K)
  cvt_f16<<<(M * C_ + 255) / 256, 256, 0, stream>>>(x, xh, M * C_);
  transpose_cvt_f16<<<(C_ * C_ + 255) / 256, 256, 0, stream>>>(Wq, Wqt, C_, C_);
  transpose_cvt_f16<<<(C_ * NKV + 255) / 256, 256, 0, stream>>>(Wk, Wkvt, C_, NKV);
  transpose_cvt_f16<<<(C_ * NKV + 255) / 256, 256, 0, stream>>>(Wv, Wkvt + (size_t)NKV * C_, C_, NKV);
  transpose_cvt_f16<<<(C_ * C_ + 255) / 256, 256, 0, stream>>>(Wp, Wpt, C_, C_);

  // Stage 1: Q and fused K|V projections (WMMA)
  gemm_xWt_f16<<<dim3(M / 64, C_ / 256), 256, 0, stream>>>(xh, Wqt, qf, M, C_, C_);
  gemm_xWt_f16<<<dim3(M / 64, NKVF / 256), 256, 0, stream>>>(xh, Wkvt, kvf, M, NKVF, C_);

  // Stage 2: RoPE + RMSNorm (q pre-scaled by 1/sqrt(64)); gated V (transposed)
  rope_rmsnorm_f16<<<M * H_, 64, 0, stream>>>(qf, C_, cosT, sinT, qhb, H_, 0.125f);
  rope_rmsnorm_f16<<<M * HKV_, 64, 0, stream>>>(kvf, NKVF, cosT, sinT, khb, HKV_, 1.0f);
  v_gate_transpose<<<M * HKV_, 64, 0, stream>>>(kvf + NKV, NKVF, x, ve, Wg, vtb);

  // Stage 3: sliding-window flash attention (WMMA)
  attn_swa<<<(B_ * H_ * (T_ / 16)) / 8, 256, 0, stream>>>(qhb, khb, vtb, yh);

  // Stage 4: output projection (WMMA) -> f32 out
  gemm_xWt_f16<<<dim3(M / 64, C_ / 256), 256, 0, stream>>>(yh, Wpt, out, M, C_, C_);
}
